// Model_39676907886421
// MI455X (gfx1250) — matrix operand order repaired and checked by reading, not compiled, not run on hardware
//
#include <hip/hip_runtime.h>
#include <math.h>

typedef _Float16 v16h __attribute__((ext_vector_type(16)));
typedef _Float16 v8h  __attribute__((ext_vector_type(8)));
typedef float    v8f  __attribute__((ext_vector_type(8)));
typedef float    v4f  __attribute__((ext_vector_type(4)));

static constexpr int Bn = 4, Tn = 2048, Dn = 128, Hn = 8, Wn = 128;
static constexpr float SCALE = 0.08838834764831845f;  // 1/sqrt(128)
static constexpr float NEGF16 = -60000.0f;            // mask sentinel, f16-safe

// Dynamic LDS layout (bytes)
static constexpr int OFF_K     = 0;                    // 160*128 f16 = 40960
static constexpr int OFF_V     = 40960;                // 40960
static constexpr int OFF_MASK  = 81920;                // 160 bytes (pad 256)
static constexpr int OFF_OB    = 82176;                // 128 f32 = 512
static constexpr int OFF_WP    = 82688;                // 16*8*128 f16 = 32768
static constexpr int OFF_BI    = 82688 + 32768;        // 32768
static constexpr int OFF_WO    = 82688;                // overlays WP in pass 2
static constexpr int OFF_SC    = 148224;               // 2*128*16*8 f16 = 65536
static constexpr int OFF_M     = 213760;               // 2*16*8 f32 = 1024
static constexpr int OFF_L     = 214784;               // 1024
static constexpr int SMEM_BYTES = 215808;

__global__ __launch_bounds__(64) void slattn_kernel(
    const float* __restrict__ q, const float* __restrict__ k, const float* __restrict__ v,
    const unsigned char* __restrict__ kpm,
    const float* __restrict__ inW,  const float* __restrict__ inB,
    const float* __restrict__ outW, const float* __restrict__ outB,
    float* __restrict__ out, float* __restrict__ attn_avg)
{
  extern __shared__ char smem[];
  _Float16*      k_lds = (_Float16*)(smem + OFF_K);
  _Float16*      v_lds = (_Float16*)(smem + OFF_V);
  unsigned char* maskb = (unsigned char*)(smem + OFF_MASK);
  float*         obias = (float*)(smem + OFF_OB);
  _Float16*      WpS   = (_Float16*)(smem + OFF_WP);
  _Float16*      BiS   = (_Float16*)(smem + OFF_BI);
  _Float16*      WoS   = (_Float16*)(smem + OFF_WO);
  _Float16*      scS   = (_Float16*)(smem + OFF_SC);
  float*         m_lds = (float*)(smem + OFF_M);
  float*         l_lds = (float*)(smem + OFF_L);

  const int tid  = threadIdx.x;
  const int lane = tid & 31;
  const int wave = tid >> 5;
  const int b    = blockIdx.x / (Tn / 32);
  const int t0   = (blockIdx.x % (Tn / 32)) * 32;
  const int tw0  = t0 + wave * 16;

  // wave32 WMMA lane-layout helpers
  const int mrow = lane & 15;               // A-row / C-column index
  const int kb   = (lane < 16) ? 0 : 8;     // A: K sub-offset (16-bit A 16x32)
  const int koff = (lane < 16) ? 0 : 16;    // unused; B is loaded with kb, same k order as A
  const int moff = (lane < 16) ? 0 : 8;     // C: M row offset (f32 C 16x16)

  // ---- stage k/v window (rows t0-127 .. t0+31, zero-padded) + mask + bias-sum
  for (int l = tid; l < 160; l += 64) {
    const int g = t0 - 127 + l;
    _Float16* kr = k_lds + l * 128;
    _Float16* vr = v_lds + l * 128;
    if (g >= 0 && g < Tn) {
      const float* ks = k + ((size_t)b * Tn + g) * Dn;
      const float* vs = v + ((size_t)b * Tn + g) * Dn;
      for (int i = 0; i < 128; i += 4) {
        v4f kf = *(const v4f*)(ks + i);
        v4f vf = *(const v4f*)(vs + i);
        #pragma unroll
        for (int j = 0; j < 4; ++j) { kr[i + j] = (_Float16)kf[j]; vr[i + j] = (_Float16)vf[j]; }
      }
      maskb[l] = kpm[(size_t)b * Tn + g] ? 1 : 0;
    } else {
      for (int i = 0; i < 128; ++i) { kr[i] = (_Float16)0.f; vr[i] = (_Float16)0.f; }
      maskb[l] = 1;
    }
  }
  for (int i = tid; i < 128; i += 64) {
    float s = 0.f;
    #pragma unroll
    for (int h = 0; h < Hn; ++h) s += outB[h * Dn + i];
    obias[i] = s;
  }

  // ---- q tile in WMMA A-layout registers (f16), 4 K-chunks of 32
  v16h qA[4];
  {
    const float* qrow = q + ((size_t)b * Tn + tw0 + mrow) * Dn;
    #pragma unroll
    for (int c = 0; c < 4; ++c) {
      const float* p = qrow + c * 32 + kb;
      v4f f0 = *(const v4f*)(p);      v4f f1 = *(const v4f*)(p + 4);
      v4f f2 = *(const v4f*)(p + 16); v4f f3 = *(const v4f*)(p + 20);
      v16h a;
      #pragma unroll
      for (int j = 0; j < 4; ++j) {
        a[j] = (_Float16)f0[j];      a[j + 4]  = (_Float16)f1[j];
        a[j + 8] = (_Float16)f2[j];  a[j + 12] = (_Float16)f3[j];
      }
      qA[c] = a;
    }
  }

  // ================= PASS 1: scores (+bias) via WMMA, running max =================
  float vm[8];
  #pragma unroll
  for (int r = 0; r < 8; ++r) vm[r] = -1.0e30f;

  for (int wg = 0; wg < Wn; wg += 16) {
    __syncthreads();
    for (int rr = tid; rr < Hn * Dn; rr += 64) {   // stage Wp, bias for 16 w's
      const int h = rr >> 7, i = rr & 127;
      const float* sw = inW + ((size_t)(h * Dn + i)) * Wn + wg;
      const float* sb = inB + ((size_t)h * (Dn * Wn)) + (size_t)i * Wn + wg;
      #pragma unroll
      for (int wm = 0; wm < 16; ++wm) {
        WpS[(wm * 8 + h) * 128 + i] = (_Float16)sw[wm];
        BiS[(wm * 8 + h) * 128 + i] = (_Float16)sb[wm];
      }
    }
    __syncthreads();

    for (int wq = 0; wq < 16; ++wq) {
      const int w = wg + wq;
      v16h B1c[4], B2c[4];
      #pragma unroll
      for (int c = 0; c < 4; ++c) {
        v16h b1 = {}; v16h b2 = {};
        if (mrow < 8) {                        // B cols 0..7 = Wp
          const _Float16* p = WpS + (wq * 8 + mrow) * 128 + c * 32 + kb;
          v8h x0 = *(const v8h*)p; v8h x1 = *(const v8h*)(p + 16);
          #pragma unroll
          for (int j = 0; j < 8; ++j) { b1[j] = x0[j]; b1[j + 8] = x1[j]; }
        } else {                               // B cols 8..15 = bias_r
          const _Float16* p = BiS + (wq * 8 + (mrow - 8)) * 128 + c * 32 + kb;
          v8h x0 = *(const v8h*)p; v8h x1 = *(const v8h*)(p + 16);
          #pragma unroll
          for (int j = 0; j < 8; ++j) { b2[j] = x0[j]; b2[j + 8] = x1[j]; }
        }
        B1c[c] = b1; B2c[c] = b2;
      }

      const _Float16* krow = k_lds + (size_t)(wave * 16 + mrow + w) * 128;
      v8f C = {};
      #pragma unroll
      for (int c = 0; c < 4; ++c) {            // C[:,0..7] += (q .* k) @ Wp
        v8h k0 = *(const v8h*)(krow + c * 32 + kb);
        v8h k1 = *(const v8h*)(krow + c * 32 + 16 + kb);
        v16h kk;
        #pragma unroll
        for (int j = 0; j < 8; ++j) { kk[j] = k0[j]; kk[j + 8] = k1[j]; }
        v16h P = qA[c] * kk;                   // v_pk_mul_f16
        C = __builtin_amdgcn_wmma_f32_16x16x32_f16(false, P, false, B1c[c], (short)0, C, false, false);
      }
      #pragma unroll
      for (int c = 0; c < 4; ++c)              // C[:,8..15] += q @ bias_r
        C = __builtin_amdgcn_wmma_f32_16x16x32_f16(false, qA[c], false, B2c[c], (short)0, C, false, false);

      const unsigned char* mb = maskb + (wave * 16 + moff + w);
      #pragma unroll
      for (int r = 0; r < 8; ++r) {
        float s = (C[r] + __shfl_xor(C[r], 8)) * SCALE;   // qkW + bias halves
        if (mb[r]) s = NEGF16;
        vm[r] = fmaxf(vm[r], s);
        if (mrow < 8)
          scS[(((size_t)wave * 128 + w) * 16 + (moff + r)) * 8 + mrow] = (_Float16)s;
      }
    }
  }

  if (mrow < 8) {
    #pragma unroll
    for (int r = 0; r < 8; ++r) m_lds[(wave * 16 + moff + r) * 8 + mrow] = vm[r];
  }
  __syncthreads();

  // ---- softmax denominators: each lane owns one t, four h
  {
    const int t = lane >> 1, h0 = (lane & 1) * 4;
    float mj[4], lj[4];
    #pragma unroll
    for (int j = 0; j < 4; ++j) { mj[j] = m_lds[(wave * 16 + t) * 8 + h0 + j]; lj[j] = 0.f; }
    for (int w = 0; w < Wn; ++w) {
      const _Float16* sp = scS + (((size_t)wave * 128 + w) * 16 + t) * 8 + h0;
      #pragma unroll
      for (int j = 0; j < 4; ++j) lj[j] += __expf((float)sp[j] - mj[j]);
    }
    #pragma unroll
    for (int j = 0; j < 4; ++j) l_lds[(wave * 16 + t) * 8 + h0 + j] = lj[j];
  }
  __syncthreads();

  // ================= PASS 2: ctx via WMMA (K=h), fused attn_avg =================
  float mv[8], il[8];
  #pragma unroll
  for (int j = 0; j < 8; ++j) {
    mv[j] = m_lds[(wave * 16 + mrow) * 8 + j];
    il[j] = 1.0f / l_lds[(wave * 16 + mrow) * 8 + j];
  }
  float acc[8][8];
  #pragma unroll
  for (int it = 0; it < 8; ++it) {
    #pragma unroll
    for (int r = 0; r < 8; ++r) acc[it][r] = 0.f;
  }

  for (int wg = 0; wg < Wn; wg += 16) {
    __syncthreads();
    for (int rr = tid; rr < Hn * Dn; rr += 64) {   // stage Wo as [w][i][h]
      const int h = rr >> 7, i = rr & 127;
      const float* sw = outW + ((size_t)(h * Dn + i)) * Wn + wg;
      #pragma unroll
      for (int wm = 0; wm < 16; ++wm) WoS[(wm * 128 + i) * 8 + h] = (_Float16)sw[wm];
    }
    __syncthreads();

    for (int wq = 0; wq < 16; ++wq) {
      const int w = wg + wq;
      v16h Bat = {};                     // B: attn[h, t]; lanes>=16 (K=8..15, 24..31) stay 0
      if (lane < 16) {
        const _Float16* sp = scS + (((size_t)wave * 128 + w) * 16 + mrow) * 8;
        float avg = 0.f;
        #pragma unroll
        for (int j = 0; j < 8; ++j) {
          float aj = __expf((float)sp[j] - mv[j]) * il[j];
          Bat[j] = (_Float16)aj;
          avg += aj;
        }
        attn_avg[((size_t)b * Tn + tw0 + mrow) * Wn + w] = avg * 0.125f;
      }
      const _Float16* vrow = v_lds + (size_t)(wave * 16 + mrow + w) * 128;
      #pragma unroll
      for (int it = 0; it < 8; ++it) {
        v16h Am = {};                    // A: Wo^T[i, h]; K=8..31 stay 0
        if (lane < 16) {
          const _Float16* wp = WoS + ((size_t)wq * 128 + it * 16 + mrow) * 8;
          v8h x = *(const v8h*)wp;
          #pragma unroll
          for (int j = 0; j < 8; ++j) Am[j] = x[j];
        }
        v8f zeroC = {};
        v8f Cw = __builtin_amdgcn_wmma_f32_16x16x32_f16(false, Am, false, Bat, (short)0, zeroC, false, false);
        v8h vv = *(const v8h*)(vrow + it * 16 + moff);
        #pragma unroll
        for (int r = 0; r < 8; ++r) acc[it][r] += Cw[r] * (float)vv[r];
      }
    }
  }

  // ---- write out = ctx + sum_h out_proj_bias
  float* orow = out + ((size_t)b * Tn + tw0 + mrow) * Dn;
  #pragma unroll
  for (int it = 0; it < 8; ++it) {
    #pragma unroll
    for (int r = 0; r < 8; ++r) {
      const int i = it * 16 + moff + r;
      orow[i] = acc[it][r] + obias[i];
    }
  }
}

extern "C" void kernel_launch(void* const* d_in, const int* in_sizes, int n_in,
                              void* d_out, int out_size, void* d_ws, size_t ws_size,
                              hipStream_t stream) {
  const float* q          = (const float*)d_in[0];
  const float* k          = (const float*)d_in[1];
  const float* v          = (const float*)d_in[2];
  const unsigned char* pm = (const unsigned char*)d_in[3];
  const float* inW        = (const float*)d_in[4];
  const float* inB        = (const float*)d_in[5];
  const float* outW       = (const float*)d_in[6];
  const float* outB       = (const float*)d_in[7];
  float* out = (float*)d_out;
  float* avg = out + (size_t)Bn * Tn * Dn;

  dim3 grid(Bn * (Tn / 32));
  dim3 block(64);
  hipLaunchKernelGGL(slattn_kernel, grid, block, SMEM_BYTES, stream,
                     q, k, v, pm, inW, inB, outW, outB, out, avg);
}
